// MultiScaleGraphConvolution_5222680232055
// MI455X (gfx1250) — matrix operand order repaired and checked by reading, not compiled, not run on hardware
//
#include <hip/hip_runtime.h>

// ---------------------------------------------------------------------------
// MultiScaleGraphConvolution for MI455X (gfx1250, wave32, WMMA)
//
// Pipeline (all on `stream`):
//   1) zero_f32      : clear aggregation buffers + counts in workspace
//   2) cvt_bf16      : convert W1/W2/Wg (f32) -> bf16 once (tensor-core diet)
//   3) edge_agg      : one wave per edge; global_atomic_add_f32 scatter into
//                      agg2 (scale 8, all 800k edges) and agg1 (scale 4,
//                      first 400k edges) + degree counts
//   4) node_fused    : one wave32 per 16-node tile; mean-normalize, then
//                      out1 = h1@W1^T+b1, out2 = h2@W2^T+b2 via
//                      v_wmma_f32_16x16x32_bf16; re-layout via LDS; gate GEMM
//                      (16x128 @ 128x64) via 16 more WMMAs; sigmoid gating;
//                      store fused output. No HBM intermediates.
// ---------------------------------------------------------------------------

typedef __attribute__((ext_vector_type(16))) __bf16 v16bf;
typedef __attribute__((ext_vector_type(8)))  float  v8f;

#define GN      100000
#define GE      800000
#define GE1     400000          // N * min(4, E/N)
#define GE2     800000          // N * min(8, E/N)
#define GD      64
#define NTILES  (GN / 16)       // 6250 exactly

__device__ __forceinline__ v8f wmma_bf16(v16bf a, v16bf b, v8f c) {
  // (neg_a, A, neg_b, B, c_mod, C, reuse_a, reuse_b)
  return __builtin_amdgcn_wmma_f32_16x16x32_bf16(false, a, false, b,
                                                 (short)0, c, false, false);
}

__device__ __forceinline__ unsigned short f2bf_bits(float f) {
  __bf16 h = (__bf16)f;                       // RNE truncation to bf16
  return __builtin_bit_cast(unsigned short, h);
}

// ---------------------------------------------------------------------------
__global__ void zero_f32(float* __restrict__ p, int n) {
  int i = blockIdx.x * blockDim.x + threadIdx.x;
  int stride = gridDim.x * blockDim.x;
  for (; i < n; i += stride) p[i] = 0.0f;
}

// Convert weights to bf16 (row-major, same shapes).
__global__ void cvt_bf16(const float* __restrict__ W1, const float* __restrict__ W2,
                         const float* __restrict__ Wg,
                         unsigned short* __restrict__ W1b,
                         unsigned short* __restrict__ W2b,
                         unsigned short* __restrict__ Wgb) {
  int i = blockIdx.x * blockDim.x + threadIdx.x;
  if (i < 64 * 64)  W1b[i] = f2bf_bits(W1[i]);
  if (i < 64 * 64)  W2b[i] = f2bf_bits(W2[i]);
  if (i < 64 * 128) Wgb[i] = f2bf_bits(Wg[i]);
}

// One wave32 per edge; lane handles feature pair (2*lane, 2*lane+1).
__global__ void edge_agg(const float* __restrict__ x, const long long* __restrict__ ei,
                         float* __restrict__ agg1, float* __restrict__ agg2,
                         float* __restrict__ c1, float* __restrict__ c2) {
  long long gid = (long long)blockIdx.x * blockDim.x + threadIdx.x;
  long long e   = gid >> 5;
  int lane      = (int)(gid & 31);
  if (e >= GE2) return;
  long long s = ei[e];         // segment id  (src row of edge_index)
  long long t = ei[GE + e];    // gathered node (tgt row)
  const float2 v = *(const float2*)(x + t * GD + lane * 2);
  float* a2 = agg2 + s * GD + lane * 2;
  atomicAdd(a2,     v.x);
  atomicAdd(a2 + 1, v.y);
  if (lane == 0) atomicAdd(c2 + s, 1.0f);
  if (e < GE1) {
    float* a1 = agg1 + s * GD + lane * 2;
    atomicAdd(a1,     v.x);
    atomicAdd(a1 + 1, v.y);
    if (lane == 0) atomicAdd(c1 + s, 1.0f);
  }
}

// ---------------------------------------------------------------------------
// Fused per-tile kernel: 8 waves / block, 1 tile (16 nodes) / wave.
__global__ void __launch_bounds__(256, 1)
node_fused(const float* __restrict__ agg1, const float* __restrict__ agg2,
           const float* __restrict__ c1,   const float* __restrict__ c2,
           const unsigned short* __restrict__ W1b,
           const unsigned short* __restrict__ W2b,
           const unsigned short* __restrict__ Wgb,
           const float* __restrict__ b1, const float* __restrict__ b2,
           const float* __restrict__ bg, float* __restrict__ out) {
  __shared__ float hcat[8][16][132];        // 132 pad: row stride % 64 != 0

  const int lane = threadIdx.x & 31;
  const int wid  = threadIdx.x >> 5;
  const int tile = blockIdx.x * 8 + wid;
  if (tile >= NTILES) return;               // whole-wave exit; no block barrier used

  const int n    = lane & 15;               // A-row m / B-col n / C-col n
  const int hi   = lane >> 4;
  const int arow = tile * 16 + n;           // node whose A-row this lane owns

  const float inv1 = 1.0f / (c1[arow] + 1e-6f);
  const float inv2 = 1.0f / (c2[arow] + 1e-6f);

  // ---- Build A tiles for h1,h2 (16x64 bf16 as 2 K-steps of 16x32) --------
  // Lane layout (16-bit A 16x32, contiguous): element j -> K = hi*16 + j
  //                                   (matches contiguous B loads).
  v16bf a1[2], a2[2];
#pragma unroll
  for (int ks = 0; ks < 2; ++ks) {
    const float* p1 = agg1 + (long long)arow * GD + ks * 32 + hi * 16;
    const float* p2 = agg2 + (long long)arow * GD + ks * 32 + hi * 16;
#pragma unroll
    for (int j = 0; j < 8; ++j) {
      a1[ks][j]     = (__bf16)(p1[j]      * inv1);
      a1[ks][8 + j] = (__bf16)(p1[8 + j] * inv1);
      a2[ks][j]     = (__bf16)(p2[j]      * inv2);
      a2[ks][8 + j] = (__bf16)(p2[8 + j] * inv2);
    }
  }

  // ---- out1 / out2 : 4 column tiles x 2 K-steps each ----------------------
  // B(k,n) = W[n][k]; lane (col n) reads 16 contiguous bf16 from W row
  // nt*16+n at K-offset ks*32 + hi*16  -> one 32B load.
  v8f o1[4], o2[4];
#pragma unroll
  for (int nt = 0; nt < 4; ++nt) {
    const __bf16* w1p = (const __bf16*)(W1b + (nt * 16 + n) * GD + hi * 16);
    const __bf16* w2p = (const __bf16*)(W2b + (nt * 16 + n) * GD + hi * 16);
    v8f acc1 = {}; v8f acc2 = {};
#pragma unroll
    for (int ks = 0; ks < 2; ++ks) {
      v16bf bw1 = *(const v16bf*)(w1p + ks * 32);
      acc1 = wmma_bf16(a1[ks], bw1, acc1);
      v16bf bw2 = *(const v16bf*)(w2p + ks * 32);
      acc2 = wmma_bf16(a2[ks], bw2, acc2);
    }
    const float bb1 = b1[nt * 16 + n];
    const float bb2 = b2[nt * 16 + n];
#pragma unroll
    for (int v = 0; v < 8; ++v) { acc1[v] += bb1; acc2[v] += bb2; }
    o1[nt] = acc1; o2[nt] = acc2;
    // D layout: element (M = v + 8*hi, N = n). Stage concat(out1,out2) in LDS
    // so it can be re-read in A layout for the gate GEMM (same wave, DS in-order).
#pragma unroll
    for (int v = 0; v < 8; ++v) {
      hcat[wid][v + hi * 8][nt * 16 + n]      = acc1[v];
      hcat[wid][v + hi * 8][64 + nt * 16 + n] = acc2[v];
    }
  }

  // ---- Gate A tiles: 16x128 bf16 = 4 K-steps of 16x32 ---------------------
  v16bf ag[4];
#pragma unroll
  for (int ks = 0; ks < 4; ++ks) {
#pragma unroll
    for (int j = 0; j < 8; ++j) {
      ag[ks][j]     = (__bf16)hcat[wid][n][ks * 32 + hi * 16 + j];
      ag[ks][8 + j] = (__bf16)hcat[wid][n][ks * 32 + hi * 16 + 8 + j];
    }
  }

  // ---- Gate GEMM + sigmoid + fuse + store --------------------------------
#pragma unroll
  for (int nt = 0; nt < 4; ++nt) {
    const __bf16* wgp = (const __bf16*)(Wgb + (nt * 16 + n) * 128 + hi * 16);
    v8f g = {};
#pragma unroll
    for (int ks = 0; ks < 4; ++ks) {
      v16bf bw = *(const v16bf*)(wgp + ks * 32);
      g = wmma_bf16(ag[ks], bw, g);
    }
    const float bbg = bg[nt * 16 + n];
#pragma unroll
    for (int v = 0; v < 8; ++v) {
      const float s = 1.0f / (1.0f + __expf(-(g[v] + bbg)));
      const float f = s * o1[nt][v] + (1.0f - s) * o2[nt][v];
      out[(long long)(tile * 16 + v + hi * 8) * GD + nt * 16 + n] = f;
    }
  }
}

// ---------------------------------------------------------------------------
extern "C" void kernel_launch(void* const* d_in, const int* in_sizes, int n_in,
                              void* d_out, int out_size, void* d_ws, size_t ws_size,
                              hipStream_t stream) {
  const float*     x  = (const float*)d_in[0];
  const long long* ei = (const long long*)d_in[1];   // int64 edge_index (2,E)
  const float*     W1 = (const float*)d_in[2];
  const float*     b1 = (const float*)d_in[3];
  const float*     W2 = (const float*)d_in[4];
  const float*     b2 = (const float*)d_in[5];
  const float*     Wg = (const float*)d_in[6];
  const float*     bg = (const float*)d_in[7];
  float*           out = (float*)d_out;

  // Workspace layout: agg1 | agg2 | c1 | c2 | W1b | W2b | Wgb  (~52 MB + 32 KB)
  float* agg1 = (float*)d_ws;
  float* agg2 = agg1 + (size_t)GN * GD;
  float* c1   = agg2 + (size_t)GN * GD;
  float* c2   = c1 + GN;
  unsigned short* W1b = (unsigned short*)(c2 + GN);
  unsigned short* W2b = W1b + 64 * 64;
  unsigned short* Wgb = W2b + 64 * 64;

  const int ztotal = 2 * GN * GD + 2 * GN;           // 13,000,000 floats
  zero_f32<<<2048, 256, 0, stream>>>(agg1, ztotal);
  cvt_bf16<<<(64 * 128 + 255) / 256, 256, 0, stream>>>(W1, W2, Wg, W1b, W2b, Wgb);

  const long long ethreads = (long long)GE2 * 32;    // 1 wave per edge
  edge_agg<<<(int)((ethreads + 255) / 256), 256, 0, stream>>>(x, ei, agg1, agg2, c1, c2);

  node_fused<<<(NTILES + 7) / 8, 256, 0, stream>>>(agg1, agg2, c1, c2,
                                                   W1b, W2b, Wgb, b1, b2, bg, out);
}
